// TransitionUp_67439576482089
// MI455X (gfx1250) — compile-verified, matrix operand order checked by reading, not run on hardware
//
#include <hip/hip_runtime.h>

typedef _Float16 v16h __attribute__((ext_vector_type(16)));
typedef float    v8f  __attribute__((ext_vector_type(8)));

constexpr int   kN    = 1048576;
constexpr int   kC    = 64;
constexpr int   kB    = 16;
constexpr float kEps  = 1e-5f;

constexpr int   kRows = 128;   // rows per block in main kernel (8 waves x 16)
constexpr int   kFStr = 65;    // padded f32 stride for X/H tile
constexpr int   kWStr = 66;    // padded f16 stride for W^T tile

// -------------------- kernel 1: zero segment-sum scratch --------------------
__global__ void k_zero(float* __restrict__ sums) {
    int i = blockIdx.x * 256 + threadIdx.x;
    if (i < kB * kC) sums[i] = 0.0f;
}

// -------------------- kernel 2: per-segment sums of x --------------------
// Segments are contiguous row ranges (cumulative end offsets in o), so each
// thread keeps a running register accumulator and only flushes (LDS atomic)
// on segment boundaries. One coalesced pass over x.
__global__ void __launch_bounds__(256)
k_segsum(const float* __restrict__ x, const int* __restrict__ o,
         float* __restrict__ sums) {
    __shared__ float part[kB][kC];
    __shared__ int   oS[kB];
    const int tid = threadIdx.x;
    for (int i = tid; i < kB * kC; i += 256) ((float*)part)[i] = 0.0f;
    if (tid < kB) oS[tid] = o[tid];
    __syncthreads();

    const int  col  = tid & 63;
    const int  rg   = tid >> 6;                 // 0..3
    const long base = (long)blockIdx.x * 1024;  // 1024 rows per block

    int cur = 0;
    {
        long row0 = base + rg;
        while (row0 >= (long)oS[cur]) cur++;
    }
    float acc = 0.0f;
    for (int i = 0; i < 256; ++i) {
        long row = base + rg + 4l * i;
        if (row >= (long)oS[cur]) {
            atomicAdd(&part[cur][col], acc);
            acc = 0.0f;
            do { cur++; } while (row >= (long)oS[cur]);
        }
        acc += x[row * 64 + col];
    }
    atomicAdd(&part[cur][col], acc);
    __syncthreads();

    for (int i = tid; i < kB * kC; i += 256) {
        float p = ((float*)part)[i];
        if (p != 0.0f) atomicAdd(&sums[i], p);
    }
}

// -------------------- kernel 3: means -> xb=relu(mean@W2+b2) -> v table ----
// v[b] = xb[b] @ W1[64:128,:] + b1   (folds the pooled half of the big GEMM
// and its bias into a 16x64 lookup table)
__global__ void k_prep(const float* __restrict__ sums, const int* __restrict__ o,
                       const float* __restrict__ W2, const float* __restrict__ b2,
                       const float* __restrict__ W1, const float* __restrict__ b1,
                       float* __restrict__ vtab) {
    __shared__ float mn[kB][kC];
    __shared__ float xb[kB][kC];
    const int j = threadIdx.x;  // 0..63
    for (int b = 0; b < kB; ++b) {
        int   c0  = (b == 0) ? 0 : o[b - 1];
        float cnt = (float)(o[b] - c0);
        mn[b][j]  = sums[b * kC + j] / cnt;
    }
    __syncthreads();
    for (int b = 0; b < kB; ++b) {
        float acc = b2[j];
        for (int k = 0; k < kC; ++k) acc += mn[b][k] * W2[k * kC + j];
        xb[b][j] = fmaxf(acc, 0.0f);
    }
    __syncthreads();
    for (int b = 0; b < kB; ++b) {
        float acc = b1[j];
        for (int k = 0; k < kC; ++k) acc += xb[b][k] * W1[(kC + k) * kC + j];
        vtab[b * kC + j] = acc;
    }
}

// -------------------- kernel 4: async-LDS + WMMA GEMM + LayerNorm + ReLU ---
// X tile is DMA'd into LDS with gfx1250 async loads (ASYNCcnt path), the
// 64x64 GEMM runs on v_wmma_f32_16x16x32_f16, and the X-tile LDS region is
// reused (via union) for the GEMM result once A-fragments are in registers.
union TileU {
    float xf[kRows][kFStr];   // staged x tile (f32)
    float hs[kRows][kFStr];   // GEMM result (f32)
};

__global__ void __launch_bounds__(256)
k_main(const float* __restrict__ x, const int* __restrict__ o,
       const float* __restrict__ W1, const float* __restrict__ vtab,
       const float* __restrict__ gamma, const float* __restrict__ beta,
       float* __restrict__ out) {
    __shared__ TileU    U;
    __shared__ _Float16 Wt[kC][kWStr];      // Wt[n][k] = W1[k][n], f16
    __shared__ float    Vl[kB][kC];         // per-segment bias vectors
    __shared__ float    muS[kRows], rsS[kRows];
    __shared__ int      segS[kRows];
    __shared__ int      oS[kB];

    const int  tid     = threadIdx.x;
    const long rowBase = (long)blockIdx.x * kRows;

    // async-DMA the 128x64 f32 x-tile straight into LDS (no VGPR round-trip)
    {
        int colv = tid & 63;
        int r0   = tid >> 6;
        #pragma unroll
        for (int i = 0; i < 32; ++i) {
            int r = r0 + 4 * i;
            unsigned long long gsrc =
                (unsigned long long)(const void*)(x + (rowBase + r) * 64 + colv);
            unsigned int ldst = (unsigned int)(unsigned long long)(const void*)&U.xf[r][colv];
            asm volatile("global_load_async_to_lds_b32 %0, %1, off"
                         :: "v"(ldst), "v"(gsrc) : "memory");
        }
    }

    if (tid < kB) oS[tid] = o[tid];
    for (int i = tid; i < kB * kC; i += 256) ((float*)Vl)[i] = vtab[i];

    // stage W1-top (64x64) transposed as f16
    {
        int n  = tid >> 2;
        int k0 = (tid & 3) * 16;
        #pragma unroll
        for (int kk = 0; kk < 16; ++kk)
            Wt[n][k0 + kk] = (_Float16)W1[(k0 + kk) * kC + n];
    }

    asm volatile("s_wait_asynccnt 0" ::: "memory");
    __syncthreads();

    // CDNA5 16-bit WMMA fragment layout:
    //   A (16x32): lane l -> row m=l&15; VGPR j holds K pair
    //              kb = (j>=4?16:0) + (l>>4)*8 + (j&3)*2
    //   B (32x16): mirrored, lane l -> col n=l&15, same K mapping
    const int wave = tid >> 5;
    const int lane = tid & 31;
    const int m    = lane & 15;
    const int hi   = lane >> 4;
    const int arow = wave * 16 + m;

    v16h a0, a1;
    #pragma unroll
    for (int j = 0; j < 8; ++j) {
        int kb = ((j >= 4) ? 16 : 0) + hi * 8 + (j & 3) * 2;
        a0[2 * j]     = (_Float16)U.xf[arow][kb];
        a0[2 * j + 1] = (_Float16)U.xf[arow][kb + 1];
        a1[2 * j]     = (_Float16)U.xf[arow][32 + kb];
        a1[2 * j + 1] = (_Float16)U.xf[arow][32 + kb + 1];
    }
    __syncthreads();   // A-fragments in registers; LDS tile is now reusable

    #pragma unroll
    for (int t = 0; t < 4; ++t) {
        const int n = t * 16 + m;
        v16h bf0, bf1;
        #pragma unroll
        for (int j = 0; j < 8; ++j) {
            int kb = ((j >= 4) ? 16 : 0) + hi * 8 + (j & 3) * 2;
            bf0[2 * j]     = Wt[n][kb];
            bf0[2 * j + 1] = Wt[n][kb + 1];
            bf1[2 * j]     = Wt[n][32 + kb];
            bf1[2 * j + 1] = Wt[n][32 + kb + 1];
        }
        v8f c = {};
        c = __builtin_amdgcn_wmma_f32_16x16x32_f16(false, a0, false, bf0,
                                                   (short)0, c, false, false);
        c = __builtin_amdgcn_wmma_f32_16x16x32_f16(false, a1, false, bf1,
                                                   (short)0, c, false, false);
        #pragma unroll
        for (int r = 0; r < 8; ++r)
            U.hs[wave * 16 + hi * 8 + r][t * 16 + m] = c[r];
    }
    __syncthreads();

    // per-row mean / rstd (padded stride -> conflict-light column walk)
    if (tid < kRows) {
        long grow = rowBase + tid;
        int  sg   = 0;
        while (grow >= (long)oS[sg]) sg++;
        segS[tid] = sg;
        float s = 0.0f, ss = 0.0f;
        #pragma unroll
        for (int cc = 0; cc < kC; ++cc) {
            float hv = U.hs[tid][cc] + Vl[sg][cc];
            s += hv; ss += hv * hv;
        }
        float mu  = s * (1.0f / kC);
        float var = ss * (1.0f / kC) - mu * mu;
        muS[tid]  = mu;
        rsS[tid]  = rsqrtf(var + kEps);
    }
    __syncthreads();

    // coalesced writeback: scale/shift + ReLU
    {
        int   colv = tid & 63;
        int   r0   = tid >> 6;
        float g    = gamma[colv];
        float bt   = beta[colv];
        #pragma unroll
        for (int i = 0; i < 32; ++i) {
            int   r  = r0 + 4 * i;
            float hv = U.hs[r][colv] + Vl[segS[r]][colv];
            float y  = (hv - muS[r]) * rsS[r] * g + bt;
            out[(rowBase + r) * 64 + colv] = fmaxf(y, 0.0f);
        }
    }
}

// -------------------- launcher --------------------
extern "C" void kernel_launch(void* const* d_in, const int* in_sizes, int n_in,
                              void* d_out, int out_size, void* d_ws, size_t ws_size,
                              hipStream_t stream) {
    const float* x     = (const float*)d_in[0];
    const int*   o     = (const int*)  d_in[1];
    const float* W2    = (const float*)d_in[2];
    const float* b2    = (const float*)d_in[3];
    const float* W1    = (const float*)d_in[4];
    const float* b1    = (const float*)d_in[5];
    const float* gamma = (const float*)d_in[6];
    const float* beta  = (const float*)d_in[7];
    float*       out   = (float*)d_out;

    float* sums = (float*)d_ws;        // [16][64]
    float* vtab = sums + kB * kC;      // [16][64]

    k_zero  <<<(kB * kC + 255) / 256, 256, 0, stream>>>(sums);
    k_segsum<<<kN / 1024,            256, 0, stream>>>(x, o, sums);
    k_prep  <<<1,                     64, 0, stream>>>(sums, o, W2, b2, W1, b1, vtab);
    k_main  <<<kN / kRows,           256, 0, stream>>>(x, o, W1, vtab, gamma, beta, out);
}
